// MatchingPursuit_31344671326403
// MI455X (gfx1250) — matrix operand order repaired and checked by reading, not compiled, not run on hardware
//
#include <hip/hip_runtime.h>

#define BATCH 1024
#define INF   512
#define OUTF  1024
#define LAMBD 0.2f
#define TOLR  1e-4f
#define MAX_ITERS 100
#define ROWS  16      // batch rows per workgroup
#define NTHREADS 256  // 8 wave32 waves

typedef __attribute__((ext_vector_type(16))) __bf16 v16bf;
typedef __attribute__((ext_vector_type(8)))  __bf16 v8bf;
typedef __attribute__((ext_vector_type(8)))  float  v8f;

__device__ inline v8f zeros8() {
  v8f z;
#pragma unroll
  for (int i = 0; i < 8; ++i) z[i] = 0.f;
  return z;
}

__device__ inline v16bf pack16(v8bf lo, v8bf hi) {
  v16bf a;
#pragma unroll
  for (int i = 0; i < 8; ++i) { a[i] = lo[i]; a[i + 8] = hi[i]; }
  return a;
}

// A fragment (16x32 bf16, row-major source, leading dim ld):
// lanes 0-15 hold row m, K = k0+0..15; lanes 16-31: K = k0+16..31
__device__ inline v16bf load_a(const __bf16* base, int ld, int m, int hk, int k0) {
  const __bf16* p = base + (size_t)m * ld + k0 + hk * 16;
  v8bf lo = *(const v8bf*)p;
  v8bf hi = *(const v8bf*)(p + 8);
  return pack16(lo, hi);
}

// B fragment (32x16 bf16) from K-major (pre-transposed) storage bt[n][k], leading dim ldk:
// lane c holds column n0+c; lanes 0-15: K = k0..k0+15, lanes 16-31: K = k0+16..k0+31
__device__ inline v16bf load_b(const __bf16* bt, int ldk, int n, int hk, int k0) {
  const __bf16* p = bt + (size_t)n * ldk + k0 + hk * 16;
  v8bf lo = *(const v8bf*)p;
  v8bf hi = *(const v8bf*)(p + 8);
  return pack16(lo, hi);
}

__device__ inline float softshrink(float x) {
  float ax = fabsf(x) - LAMBD;
  ax = ax > 0.f ? ax : 0.f;
  return x > 0.f ? ax : -ax;
}

// ---------------- prep: f32 -> bf16 (+ transposes for K-major B operands) ----------------
__global__ void prep_xw(const float* __restrict__ x, const float* __restrict__ w,
                        __bf16* __restrict__ xb, __bf16* __restrict__ wb) {
  int i = blockIdx.x * blockDim.x + threadIdx.x;
  if (i < BATCH * INF) { xb[i] = (__bf16)x[i]; wb[i] = (__bf16)w[i]; }
}
__global__ void prep_wt(const float* __restrict__ w, __bf16* __restrict__ wtb) {
  int i = blockIdx.x * blockDim.x + threadIdx.x;
  if (i < INF * OUTF) { int n = i / OUTF, k = i % OUTF; wtb[i] = (__bf16)w[(size_t)k * INF + n]; }
}
__global__ void prep_mt(const float* __restrict__ Mi, __bf16* __restrict__ mtb) {
  int i = blockIdx.x * blockDim.x + threadIdx.x;
  if (i < OUTF * OUTF) { int n = i >> 10, k = i & 1023; mtb[i] = (__bf16)Mi[(size_t)k * OUTF + n]; }
}

// ---------------- persistent fused ADMM solver: one WG owns 16 batch rows ----------------
__global__ __launch_bounds__(NTHREADS, 1) void admm_persistent(
    const __bf16* __restrict__ xb,   // x  bf16      (BATCH x INF) row-major
    const __bf16* __restrict__ wb,   // W  bf16      (OUTF x INF)  row-major  -> B for A_dot_b
    const __bf16* __restrict__ wtb,  // Wᵀ bf16      (INF x OUTF)  K-major    -> B for decode
    const __bf16* __restrict__ mtb,  // M_invᵀ bf16  (OUTF x OUTF) K-major    -> B for ADMM
    float* __restrict__ enc, float* __restrict__ dec) {
  __shared__ __align__(16) float  s_adotb[ROWS * OUTF];  // 64 KB, written once
  __shared__ __align__(16) float  s_v[ROWS * OUTF];      // 64 KB
  __shared__ __align__(16) float  s_u[ROWS * OUTF];      // 64 KB
  __shared__ __align__(16) __bf16 s_beff[ROWS * OUTF];   // 32 KB
  __shared__ __align__(16) __bf16 s_vsol[ROWS * OUTF];   // 32 KB  (total 256 KB <= 320 KB WGP LDS)
  __shared__ float s_dx2[ROWS], s_x2[ROWS];
  __shared__ int s_solved[ROWS], s_newly[ROWS], s_active[ROWS];

  const int tid = threadIdx.x, lane = tid & 31, wave = tid >> 5;
  const int m = lane & 15, hk = lane >> 4;
  const int row0 = blockIdx.x * ROWS;

  for (int i = tid; i < ROWS * OUTF; i += NTHREADS) {
    s_v[i] = 0.f; s_u[i] = 0.f; s_vsol[i] = (__bf16)0.f;
  }
  if (tid < ROWS) s_solved[tid] = 0;

  // ---- A_dot_b = x @ Wᵀ (16 x 1024 per WG) -> LDS ----
  {
    v8f acc[8];
#pragma unroll
    for (int t = 0; t < 8; ++t) acc[t] = zeros8();
    for (int k0 = 0; k0 < INF; k0 += 32) {
      v16bf a = load_a(xb + (size_t)row0 * INF, INF, m, hk, k0);
#pragma unroll
      for (int t = 0; t < 8; ++t) {
        int n0 = wave * 128 + t * 16;
        v16bf b = load_b(wb, INF, n0 + m, hk, k0);
        acc[t] = __builtin_amdgcn_wmma_f32_16x16x32_bf16(false, a, false, b, (short)0,
                                                         acc[t], false, false);
      }
    }
#pragma unroll
    for (int t = 0; t < 8; ++t) {
      int col = wave * 128 + t * 16 + m;
#pragma unroll
      for (int r = 0; r < 8; ++r) {
        int row = r + hk * 8;
        s_adotb[row * OUTF + col] = acc[t][r];
      }
    }
  }
  __syncthreads();

  // ---- 100 fused ADMM iterations, all state LDS-resident ----
  v8f acc[8];
  for (int it = 0; it < MAX_ITERS; ++it) {
    for (int i = tid; i < ROWS * OUTF; i += NTHREADS)
      s_beff[i] = (__bf16)(s_adotb[i] + s_v[i] - s_u[i]);
    if (tid < ROWS) { s_dx2[tid] = 0.f; s_x2[tid] = 0.f; }
    __syncthreads();

#pragma unroll
    for (int t = 0; t < 8; ++t) acc[t] = zeros8();
    for (int k0 = 0; k0 < OUTF; k0 += 32) {
      // pull next K-block of this wave's M_invT rows toward the WGP$ (global_prefetch_b8)
      if (k0 + 32 < OUTF)
        __builtin_prefetch(mtb + (size_t)(wave * 128 + m) * OUTF + k0 + 32 + hk * 16, 0, 1);
      v16bf a = load_a(s_beff, OUTF, m, hk, k0);  // LDS-resident A, reused by 8 tiles
#pragma unroll
      for (int t = 0; t < 8; ++t) {
        int n0 = wave * 128 + t * 16;
        v16bf b = load_b(mtb, OUTF, n0 + m, hk, k0);  // L2-resident M_invT
        acc[t] = __builtin_amdgcn_wmma_f32_16x16x32_bf16(false, a, false, b, (short)0,
                                                         acc[t], false, false);
      }
    }
    // per-row norms: register accumulate over tiles, shfl-reduce across the 16 lanes
    // sharing each row, single ds_add_f32 per row per half-wave
#pragma unroll
    for (int r = 0; r < 8; ++r) {
      int row = r + hk * 8;
      float dx2 = 0.f, x2 = 0.f;
#pragma unroll
      for (int t = 0; t < 8; ++t) {
        int idx = row * OUTF + wave * 128 + t * 16 + m;
        float xk = acc[t][r];
        float vn = softshrink(xk + s_u[idx]);
        float d = vn - s_v[idx];
        dx2 += d * d;
        x2 += vn * vn;
      }
#pragma unroll
      for (int s = 8; s >= 1; s >>= 1) {
        dx2 += __shfl_xor(dx2, s, 16);
        x2  += __shfl_xor(x2, s, 16);
      }
      if (m == 0) {
        atomicAdd(&s_dx2[row], dx2);
        atomicAdd(&s_x2[row], x2);
      }
    }
    __syncthreads();
    if (tid < ROWS) {
      bool conv = sqrtf(s_dx2[tid]) < (TOLR * sqrtf(s_x2[tid]));  // 0/0 -> false, as ref
      int newly = (conv && !s_solved[tid]) ? 1 : 0;
      s_newly[tid] = newly;
      if (newly) s_solved[tid] = 1;
      s_active[tid] = s_solved[tid] ? 0 : 1;
    }
    __syncthreads();
    // commit: frozen rows keep old v/u; newly converged rows capture v_sol
#pragma unroll
    for (int r = 0; r < 8; ++r) {
      int row = r + hk * 8;
      int nw = s_newly[row], act = s_active[row];
#pragma unroll
      for (int t = 0; t < 8; ++t) {
        int idx = row * OUTF + wave * 128 + t * 16 + m;
        float xk = acc[t][r];
        float uu = s_u[idx];
        float vn = softshrink(xk + uu);
        if (nw) s_vsol[idx] = (__bf16)vn;
        if (act) { s_v[idx] = vn; s_u[idx] = uu + xk - vn; }
      }
    }
    __syncthreads();
  }

  // ---- encoded out (zeros remain for never-converged rows) ----
  for (int i = tid; i < ROWS * OUTF; i += NTHREADS)
    enc[(size_t)row0 * OUTF + i] = (float)s_vsol[i];

  // ---- decoded = v_sol @ W ----
  {
    v8f dacc[4];
#pragma unroll
    for (int t = 0; t < 4; ++t) dacc[t] = zeros8();
    for (int k0 = 0; k0 < OUTF; k0 += 32) {
      v16bf a = load_a(s_vsol, OUTF, m, hk, k0);
#pragma unroll
      for (int t = 0; t < 4; ++t) {
        int n0 = wave * 64 + t * 16;
        v16bf b = load_b(wtb, OUTF, n0 + m, hk, k0);
        dacc[t] = __builtin_amdgcn_wmma_f32_16x16x32_bf16(false, a, false, b, (short)0,
                                                          dacc[t], false, false);
      }
    }
#pragma unroll
    for (int t = 0; t < 4; ++t) {
      int col = wave * 64 + t * 16 + m;
#pragma unroll
      for (int r = 0; r < 8; ++r) {
        int row = r + hk * 8;
        dec[(size_t)(row0 + row) * INF + col] = dacc[t][r];
      }
    }
  }
}

extern "C" void kernel_launch(void* const* d_in, const int* in_sizes, int n_in,
                              void* d_out, int out_size, void* d_ws, size_t ws_size,
                              hipStream_t stream) {
  (void)in_sizes; (void)n_in; (void)out_size; (void)ws_size;
  const float* x  = (const float*)d_in[0];
  const float* w  = (const float*)d_in[1];
  const float* Mi = (const float*)d_in[2];

  char* ws = (char*)d_ws;
  __bf16* xb  = (__bf16*)(ws);                               // 1 MB
  __bf16* wb  = (__bf16*)(ws + (size_t)BATCH * INF * 2);     // 1 MB
  __bf16* wtb = (__bf16*)(ws + (size_t)BATCH * INF * 2 * 2); // 1 MB
  __bf16* mtb = (__bf16*)(ws + (size_t)BATCH * INF * 2 * 3); // 2 MB

  float* enc = (float*)d_out;
  float* dec = enc + (size_t)BATCH * OUTF;

  prep_xw<<<(BATCH * INF + 255) / 256, 256, 0, stream>>>(x, w, xb, wb);
  prep_wt<<<(INF * OUTF + 255) / 256, 256, 0, stream>>>(w, wtb);
  prep_mt<<<(OUTF * OUTF + 255) / 256, 256, 0, stream>>>(Mi, mtb);
  admm_persistent<<<BATCH / ROWS, NTHREADS, 0, stream>>>(xb, wb, wtb, mtb, enc, dec);
}
